// CoAttentionMessagePassingNetwork_51024211476720
// MI455X (gfx1250) — matrix operand order repaired and checked by reading, not compiled, not run on hardware
//
#include <hip/hip_runtime.h>
#include <hip/hip_bf16.h>
#include <math.h>
#include <stdint.h>

// ---------------- problem constants (match reference) ----------------
#define B_    32
#define NPG_  64
#define N_    2048          // nodes per side
#define DEG_  16
#define E_    32768         // edges per side
#define P_    131072        // cross-graph pairs
#define D_    128
#define DR_   256
#define STEPS_ 3
#define INV_TEMP_ 0.08838834764831845f   // 1/sqrt(128)
#define LN_EPS_ 1e-5f

typedef __attribute__((ext_vector_type(16))) __bf16 bf16x16;
typedef __attribute__((ext_vector_type(8)))  __bf16 bf16x8;
typedef __attribute__((ext_vector_type(8)))  float  f32x8;
typedef __attribute__((ext_vector_type(4)))  float  f32x4;
typedef int v4i __attribute__((vector_size(16)));   // matches builtin param type

#if __has_builtin(__builtin_amdgcn_global_load_async_to_lds_b128)
#define USE_ASYNC_LDS 1
typedef __attribute__((address_space(1))) v4i* gvec_ptr;
typedef __attribute__((address_space(3))) v4i* lvec_ptr;
#else
#define USE_ASYNC_LDS 0
#endif

// =====================================================================
// Weight pre-pack: Wt[n*K + k] = bf16(W[k*N + n])  (column-major bf16)
// Done once per call so GEMM blocks can DMA a contiguous 32KB strip.
// =====================================================================
__global__ void pack_wt_kernel(const float* __restrict__ W, __bf16* __restrict__ Wt,
                               int N, int K)
{
    const int i = blockIdx.x * blockDim.x + threadIdx.x;
    if (i >= N * K) return;
    const int k = i / N;
    const int n = i - k * N;
    Wt[(size_t)n * K + k] = (__bf16)W[i];
}

// =====================================================================
// Fast path: C = act(A[M,K] @ W[K,N] + bias), pre-packed bf16 W^T in LDS.
//   - block = 256 threads (8 waves); blockIdx.y selects a 128-column strip.
//   - weight strip is contiguous -> async global->LDS B128 DMA (ASYNCcnt).
//   - each wave register-blocks 16x128: one A fragment -> 8 WMMAs/K-chunk.
//   - ABF16: A stored as packed bf16 (2 global b128/chunk, no cvt)
//   - OBF16: C stored as bf16 (chained-GEMM intermediates)
// =====================================================================
template <bool ABF16, bool OBF16>
__global__ __launch_bounds__(256)
void wmma_gemm_wlds_kernel(const void* __restrict__ Av, const __bf16* __restrict__ Wt,
                           const float* __restrict__ bias, void* __restrict__ Cv,
                           int M, int N, int K, int act)
{
    extern __shared__ __bf16 Wlds[];   // [128 cols][K] column-major, K*128*2 bytes
    const int nbase = blockIdx.y << 7;

    // ---- stage the contiguous 128-column weight strip into LDS ----
    const __bf16* wsrc = Wt + (size_t)nbase * K;
    const int nchunks = K << 4;        // K*128*2 / 16 bytes
#if USE_ASYNC_LDS
    {
        const unsigned long long gbase = (unsigned long long)(uintptr_t)(const void*)wsrc;
        const unsigned lofs = (unsigned)(uintptr_t)(void*)Wlds;
        for (int c = threadIdx.x; c < nchunks; c += 256) {
            __builtin_amdgcn_global_load_async_to_lds_b128(
                (gvec_ptr)(uintptr_t)(gbase + ((unsigned long long)c << 4)),
                (lvec_ptr)(uintptr_t)(lofs + ((unsigned)c << 4)),
                0, 0);
        }
#if __has_builtin(__builtin_amdgcn_s_wait_asynccnt)
        __builtin_amdgcn_s_wait_asynccnt(0);
#else
        asm volatile("s_wait_asynccnt 0x0" ::: "memory");
#endif
    }
#else
    for (int c = threadIdx.x; c < nchunks; c += 256)
        ((uint4*)Wlds)[c] = ((const uint4*)wsrc)[c];
#endif
    __syncthreads();

    const int wave = threadIdx.x >> 5;
    const int lane = threadIdx.x & 31;
    const int tm   = blockIdx.x * 8 + wave;
    if (tm >= (M >> 4)) return;                 // whole wave exits together

    const int half = lane >> 4;
    const int l16  = lane & 15;

    f32x8 acc[8];
#pragma unroll
    for (int t = 0; t < 8; ++t) acc[t] = {};

    for (int kk = 0; kk < K; kk += 32) {
        // ---- A fragment: K = g*16 + half*8 + 2r + e  -> a[8g + (2r+e)] ----
        bf16x16 a;
        if constexpr (ABF16) {
            const __bf16* ap = (const __bf16*)Av + (size_t)(tm * 16 + l16) * K
                               + kk + (half << 3);
#pragma unroll
            for (int g = 0; g < 2; ++g) {
                const bf16x8 f = *(const bf16x8*)(ap + (g << 4));
#pragma unroll
                for (int j = 0; j < 8; ++j) a[(g << 3) + j] = f[j];
            }
        } else {
            const float* ap = (const float*)Av + (size_t)(tm * 16 + l16) * K
                              + kk + (half << 3);
#pragma unroll
            for (int g = 0; g < 2; ++g) {
                const f32x4 f0 = *(const f32x4*)(ap + (g << 4));
                const f32x4 f1 = *(const f32x4*)(ap + (g << 4) + 4);
#pragma unroll
                for (int j = 0; j < 4; ++j) {
                    a[(g << 3) + j]     = (__bf16)f0[j];
                    a[(g << 3) + 4 + j] = (__bf16)f1[j];
                }
            }
        }
        // ---- 8 n-tiles: B fragment is two bf16x8 runs (k = half*8, 16 + half*8) in LDS ----
#pragma unroll
        for (int t = 0; t < 8; ++t) {
            const __bf16* bp = Wlds + (size_t)((t << 4) + l16) * K + kk + (half << 3);
            const bf16x8 b0 = *(const bf16x8*)(bp);
            const bf16x8 b1 = *(const bf16x8*)(bp + 16);
            bf16x16 b;
#pragma unroll
            for (int j = 0; j < 8; ++j) { b[j] = b0[j]; b[8 + j] = b1[j]; }
            acc[t] = __builtin_amdgcn_wmma_f32_16x16x32_bf16(
                         false, a, false, b, (short)0, acc[t], false, false);
        }
    }

#pragma unroll
    for (int t = 0; t < 8; ++t) {
        const int ncol = nbase + (t << 4) + l16;
        const float bv = bias ? bias[ncol] : 0.0f;
#pragma unroll
        for (int r = 0; r < 8; ++r) {
            const int m = tm * 16 + (half << 3) + r;
            float v = acc[t][r] + bv;
            if (act) v = v > 0.0f ? v : 0.01f * v;     // LeakyReLU(0.01)
            if constexpr (OBF16)
                ((__bf16*)Cv)[(size_t)m * N + ncol] = (__bf16)v;
            else
                ((float*)Cv)[(size_t)m * N + ncol] = v;
        }
    }
}

// =====================================================================
// Generic strided / batched GEMM (small attention GEMMs):
//   C[b][m][n] = sum_k A(b,m,k) * B(b,k,n)
// =====================================================================
__global__ __launch_bounds__(256)
void wmma_gemm_kernel(const float* __restrict__ A, const float* __restrict__ Bm,
                      float* __restrict__ C,
                      int M, int N, int K,
                      int saM, int saK, int batchA,
                      int sbK, int sbN, int batchB,
                      int batchC)
{
    const int wave = threadIdx.x >> 5;
    const int lane = threadIdx.x & 31;
    const int tilesN = N >> 4;
    const int tilesM = M >> 4;
    const int tid = blockIdx.x * (blockDim.x >> 5) + wave;
    if (tid >= tilesM * tilesN) return;

    const int tm = tid / tilesN;
    const int tn = tid - tm * tilesN;

    const float* Ab = A  + (size_t)blockIdx.z * batchA;
    const float* Bb = Bm + (size_t)blockIdx.z * batchB;
    float*       Cb = C  + (size_t)blockIdx.z * batchC;

    const int half = lane >> 4;
    const int l16  = lane & 15;
    const int mrow = tm * 16 + l16;
    const int ncol = tn * 16 + l16;

    f32x8 c = {};
    for (int kk = 0; kk < K; kk += 32) {
        bf16x16 a, b;
#pragma unroll
        for (int v = 0; v < 8; ++v) {
            const int ka = kk + ((v >> 2) << 4) + (half << 3) + ((v & 3) << 1);
            a[2*v]   = (__bf16)Ab[(size_t)mrow * saM + (size_t)(ka)     * saK];
            a[2*v+1] = (__bf16)Ab[(size_t)mrow * saM + (size_t)(ka + 1) * saK];
            const int kb = kk + ((v >> 2) << 4) + (half << 3) + ((v & 3) << 1);
            b[2*v]   = (__bf16)Bb[(size_t)(kb)     * sbK + (size_t)ncol * sbN];
            b[2*v+1] = (__bf16)Bb[(size_t)(kb + 1) * sbK + (size_t)ncol * sbN];
        }
        c = __builtin_amdgcn_wmma_f32_16x16x32_bf16(
                false, a, false, b, (short)0, c, false, false);
    }
#pragma unroll
    for (int r = 0; r < 8; ++r) {
        const int m = tm * 16 + (half << 3) + r;
        Cb[(size_t)m * N + ncol] = c[r];
    }
}

// ---------------------------------------------------------------------
// im[n,d] = sum_{t<16} nh[idx_j[16n+t], d] * eh[16n+t, d]
// ---------------------------------------------------------------------
__global__ void edge_aggregate_kernel(const float* __restrict__ nh,
                                      const float* __restrict__ eh,
                                      const int*   __restrict__ idxj,
                                      float* __restrict__ im)
{
    const int idx = blockIdx.x * blockDim.x + threadIdx.x;
    if (idx >= N_ * D_) return;
    const int n = idx >> 7;
    const int d = idx & 127;
    const size_t eb = (size_t)n * DEG_;
    float acc = 0.0f;
#pragma unroll
    for (int t = 0; t < DEG_; ++t) {
        const int j = idxj[eb + t];
        acc += nh[((size_t)j << 7) + d] * eh[((eb + t) << 7) + d];
    }
    im[idx] = acc;
}

// ---------------------------------------------------------------------
// segment softmax: one wave per row (b,i); 64 logits, temp = sqrt(D)
// ---------------------------------------------------------------------
__global__ void softmax_rows_kernel(const float* __restrict__ sc, float* __restrict__ a)
{
    const int wid  = (blockIdx.x * blockDim.x + threadIdx.x) >> 5;
    const int lane = threadIdx.x & 31;
    if (wid >= B_ * NPG_) return;
    const int b = wid >> 6, i = wid & 63;
    const float* row = sc + ((size_t)b << 12) + ((size_t)i << 6);
    float x0 = row[lane], x1 = row[lane + 32];
    float m = fmaxf(x0, x1);
    for (int o = 16; o; o >>= 1) m = fmaxf(m, __shfl_xor(m, o, 32));
    float e0 = __expf((x0 - m) * INV_TEMP_);
    float e1 = __expf((x1 - m) * INV_TEMP_);
    float s = e0 + e1;
    for (int o = 16; o; o >>= 1) s += __shfl_xor(s, o, 32);
    const float inv = 1.0f / (s + 1e-8f);
    float* out = a + ((size_t)b << 12) + ((size_t)i << 6);
    out[lane] = e0 * inv;
    out[lane + 32] = e1 * inv;
}

__global__ void softmax_cols_kernel(const float* __restrict__ sc, float* __restrict__ a)
{
    const int wid  = (blockIdx.x * blockDim.x + threadIdx.x) >> 5;
    const int lane = threadIdx.x & 31;
    if (wid >= B_ * NPG_) return;
    const int b = wid >> 6, j = wid & 63;
    const float* base = sc + ((size_t)b << 12) + j;
    float x0 = base[(size_t)lane << 6];
    float x1 = base[(size_t)(lane + 32) << 6];
    float m = fmaxf(x0, x1);
    for (int o = 16; o; o >>= 1) m = fmaxf(m, __shfl_xor(m, o, 32));
    float e0 = __expf((x0 - m) * INV_TEMP_);
    float e1 = __expf((x1 - m) * INV_TEMP_);
    float s = e0 + e1;
    for (int o = 16; o; o >>= 1) s += __shfl_xor(s, o, 32);
    const float inv = 1.0f / (s + 1e-8f);
    float* outb = a + ((size_t)b << 12) + j;
    outb[(size_t)lane << 6]        = e0 * inv;
    outb[(size_t)(lane + 32) << 6] = e1 * inv;
}

// ---------------------------------------------------------------------
// node = LayerNorm(node + im + om) * g + b    (one wave per 128-row)
// ---------------------------------------------------------------------
__global__ void residual_ln_kernel(float* __restrict__ node,
                                   const float* __restrict__ im,
                                   const float* __restrict__ om,
                                   const float* __restrict__ g,
                                   const float* __restrict__ b)
{
    const int row  = blockIdx.x * blockDim.y + threadIdx.y;
    const int lane = threadIdx.x;
    if (row >= N_) return;
    const size_t base = (size_t)row << 7;
    float x[4];
    float s = 0.0f;
#pragma unroll
    for (int q = 0; q < 4; ++q) {
        const int d = lane + (q << 5);
        x[q] = node[base + d] + im[base + d] + om[base + d];
        s += x[q];
    }
    for (int o = 16; o; o >>= 1) s += __shfl_xor(s, o, 32);
    const float mu = s * (1.0f / 128.0f);
    float v = 0.0f;
#pragma unroll
    for (int q = 0; q < 4; ++q) { const float t = x[q] - mu; v += t * t; }
    for (int o = 16; o; o >>= 1) v += __shfl_xor(v, o, 32);
    const float inv = rsqrtf(v * (1.0f / 128.0f) + LN_EPS_);
#pragma unroll
    for (int q = 0; q < 4; ++q) {
        const int d = lane + (q << 5);
        node[base + d] = (x[q] - mu) * inv * g[d] + b[d];
    }
}

// g[b,d] = sum_{i<64} h[b*64+i, d]
__global__ void graph_sum_kernel(const float* __restrict__ h, float* __restrict__ g)
{
    const int idx = blockIdx.x * blockDim.x + threadIdx.x;
    if (idx >= B_ * DR_) return;
    const int b = idx / DR_;
    const int d = idx - b * DR_;
    float s = 0.0f;
    for (int i = 0; i < NPG_; ++i)
        s += h[((size_t)(b * NPG_ + i)) * DR_ + d];
    g[idx] = s;
}

// =====================================================================
// host-side helpers
// =====================================================================
template <bool ABF16, bool OBF16>
static void gemmW(hipStream_t st, const void* A, const __bf16* Wt, const float* bias,
                  void* C, int M, int N, int K, int act)
{
    dim3 grid(((M >> 4) + 7) / 8, N >> 7, 1);
    const size_t lds = (size_t)K * 128 * sizeof(__bf16);
    wmma_gemm_wlds_kernel<ABF16, OBF16><<<grid, dim3(256), lds, st>>>(
        A, Wt, bias, C, M, N, K, act);
}

static void gemmG(hipStream_t st, const float* A, const float* Bm, float* C,
                  int M, int N, int K,
                  int saM, int saK, int batchA,
                  int sbK, int sbN, int batchB,
                  int batchC, int batch)
{
    const int tiles = (M >> 4) * (N >> 4);
    dim3 grid((tiles + 7) / 8, 1, batch);
    wmma_gemm_kernel<<<grid, dim3(256), 0, st>>>(A, Bm, C, M, N, K,
                                                 saM, saK, batchA,
                                                 sbK, sbN, batchB, batchC);
}

extern "C" void kernel_launch(void* const* d_in, const int* in_sizes, int n_in,
                              void* d_out, int out_size, void* d_ws, size_t ws_size,
                              hipStream_t stream)
{
    (void)in_sizes; (void)n_in; (void)out_size; (void)ws_size;

    const float* node1_in = (const float*)d_in[0];
    const float* edge1    = (const float*)d_in[1];
    const float* node2_in = (const float*)d_in[2];
    const float* edge2    = (const float*)d_in[3];
    const int*   idxj1    = (const int*)d_in[7];
    const int*   idxj2    = (const int*)d_in[9];
    const int PB = 14;
    auto F = [&](int i) { return (const float*)d_in[i]; };

    float* dout = (float*)d_out;
    float* g1o  = dout;                         // [32,256]
    float* g2o  = dout + B_ * DR_;              // [32,256]
    float* a1o  = dout + 2 * B_ * DR_;          // [P]
    float* a2o  = a1o + P_;                     // [P]

    // ---- workspace carve-up (floats) ----
    float* ws = (float*)d_ws;
    size_t o = 0;
    __bf16* et16 = (__bf16*)(ws + o); o += (size_t)E_ * D_ / 2;  // bf16 edge hidden
    float* ehb  = ws + o; o += (size_t)E_ * D_;
    float* n1   = ws + o; o += (size_t)N_ * D_;
    float* n2   = ws + o; o += (size_t)N_ * D_;
    float* nh   = ws + o; o += (size_t)N_ * D_;
    float* k1   = ws + o; o += (size_t)N_ * D_;
    float* k2   = ws + o; o += (size_t)N_ * D_;
    float* v1   = ws + o; o += (size_t)N_ * D_;
    float* v2   = ws + o; o += (size_t)N_ * D_;
    float* sc   = ws + o; o += (size_t)P_;
    float* wa1  = ws + o; o += (size_t)P_;
    float* wa2  = ws + o; o += (size_t)P_;
    float* im1  = ws + o; o += (size_t)N_ * D_;
    float* im2  = ws + o; o += (size_t)N_ * D_;
    float* mr1  = ws + o; o += (size_t)N_ * D_;
    float* mr2  = ws + o; o += (size_t)N_ * D_;
    float* om1  = ws + o; o += (size_t)N_ * D_;
    float* om2  = ws + o; o += (size_t)N_ * D_;
    float* hbuf = ws + o; o += (size_t)N_ * DR_;
    __bf16* wtb = (__bf16*)(ws + o); o += (18 * 16384 + 32768) / 2;  // packed W^T

    (void)hipMemcpyAsync(n1, node1_in, (size_t)N_ * D_ * sizeof(float),
                         hipMemcpyDeviceToDevice, stream);
    (void)hipMemcpyAsync(n2, node2_in, (size_t)N_ * D_ * sizeof(float),
                         hipMemcpyDeviceToDevice, stream);

    // ---- pre-pack all weights to column-major bf16 (once per call) ----
    const int wsel[6] = {0, 1, 3, 5, 6, 7};   // Wn,We1,We2,Wk,Wv,Wo within step params
    for (int s = 0; s < STEPS_; ++s)
        for (int w = 0; w < 6; ++w)
            pack_wt_kernel<<<dim3(64), dim3(256), 0, stream>>>(
                F(PB + s * 11 + wsel[w]), wtb + (size_t)(s * 6 + w) * 16384, 128, 128);
    pack_wt_kernel<<<dim3(128), dim3(256), 0, stream>>>(
        F(PB + 33), wtb + 18 * 16384, 256, 128);
    const __bf16* WtRo = wtb + 18 * 16384;

    const dim3 aggGrid((N_ * D_ + 255) / 256), blk256(256);
    const dim3 smGrid((B_ * NPG_ * 32 + 255) / 256);
    const dim3 lnGrid(N_ / 8), lnBlk(32, 8);

    for (int s = 0; s < STEPS_; ++s) {
        const int po = PB + s * 11;
        const __bf16* WtWn  = wtb + (size_t)(s * 6 + 0) * 16384;
        const __bf16* WtWe1 = wtb + (size_t)(s * 6 + 1) * 16384;
        const __bf16* WtWe2 = wtb + (size_t)(s * 6 + 2) * 16384;
        const __bf16* WtWk  = wtb + (size_t)(s * 6 + 3) * 16384;
        const __bf16* WtWv  = wtb + (size_t)(s * 6 + 4) * 16384;
        const __bf16* WtWo  = wtb + (size_t)(s * 6 + 5) * 16384;
        const float* be1 = F(po + 2);
        const float* be2 = F(po + 4);
        const float* bo  = F(po + 8);
        const float* lng = F(po + 9);
        const float* lnb = F(po + 10);
        float* a1p = (s == STEPS_ - 1) ? a1o : wa1;
        float* a2p = (s == STEPS_ - 1) ? a2o : wa2;

        // ---- message passing, side 1 ----
        gemmW<false, true >(stream, edge1, WtWe1, be1, et16, E_, D_, D_, 1);
        gemmW<true , false>(stream, et16,  WtWe2, be2, ehb,  E_, D_, D_, 1);
        gemmW<false, false>(stream, n1,    WtWn, nullptr, nh, N_, D_, D_, 0);
        edge_aggregate_kernel<<<aggGrid, blk256, 0, stream>>>(nh, ehb, idxj1, im1);
        // ---- message passing, side 2 (reuse et16/ehb/nh) ----
        gemmW<false, true >(stream, edge2, WtWe1, be1, et16, E_, D_, D_, 1);
        gemmW<true , false>(stream, et16,  WtWe2, be2, ehb,  E_, D_, D_, 1);
        gemmW<false, false>(stream, n2,    WtWn, nullptr, nh, N_, D_, D_, 0);
        edge_aggregate_kernel<<<aggGrid, blk256, 0, stream>>>(nh, ehb, idxj2, im2);

        // ---- co-attention ----
        gemmW<false, false>(stream, n1, WtWk, nullptr, k1, N_, D_, D_, 0);
        gemmW<false, false>(stream, n2, WtWk, nullptr, k2, N_, D_, D_, 0);
        gemmW<false, false>(stream, n1, WtWv, nullptr, v1, N_, D_, D_, 0);
        gemmW<false, false>(stream, n2, WtWv, nullptr, v2, N_, D_, D_, 0);
        // scores_b = K1_b (64x128) @ K2_b^T (128x64), batched over 32 graphs
        gemmG(stream, k1, k2, sc, NPG_, NPG_, D_,
              D_, 1, NPG_ * D_,
              1, D_, NPG_ * D_,
              NPG_ * NPG_, B_);
        softmax_rows_kernel<<<smGrid, blk256, 0, stream>>>(sc, a1p);
        softmax_cols_kernel<<<smGrid, blk256, 0, stream>>>(sc, a2p);
        // msg1_b = A1_b (64x64) @ V2_b (64x128)
        gemmG(stream, a1p, v2, mr1, NPG_, D_, NPG_,
              NPG_, 1, NPG_ * NPG_,
              D_, 1, NPG_ * D_,
              NPG_ * D_, B_);
        // msg2_b = A2_b^T (64x64) @ V1_b (64x128)
        gemmG(stream, a2p, v1, mr2, NPG_, D_, NPG_,
              1, NPG_, NPG_ * NPG_,
              D_, 1, NPG_ * D_,
              NPG_ * D_, B_);
        gemmW<false, false>(stream, mr1, WtWo, bo, om1, N_, D_, D_, 1);
        gemmW<false, false>(stream, mr2, WtWo, bo, om2, N_, D_, D_, 1);

        residual_ln_kernel<<<lnGrid, lnBlk, 0, stream>>>(n1, im1, om1, lng, lnb);
        residual_ln_kernel<<<lnGrid, lnBlk, 0, stream>>>(n2, im2, om2, lng, lnb);
    }

    // ---- readout ----
    const float* rob = F(PB + 34);
    gemmW<false, false>(stream, n1, WtRo, rob, hbuf, N_, DR_, D_, 1);
    graph_sum_kernel<<<dim3((B_ * DR_ + 255) / 256), blk256, 0, stream>>>(hbuf, g1o);
    gemmW<false, false>(stream, n2, WtRo, rob, hbuf, N_, DR_, D_, 1);
    graph_sum_kernel<<<dim3((B_ * DR_ + 255) / 256), blk256, 0, stream>>>(hbuf, g2o);
}
